// GCN_10050223473070
// MI455X (gfx1250) — compile-verified, matrix operand order checked by reading, not run on hardware
//
#include <hip/hip_runtime.h>
#include <hip/hip_bf16.h>

typedef __attribute__((ext_vector_type(16))) _Float16 v16h;
typedef __attribute__((ext_vector_type(8)))  _Float16 h8;
typedef __attribute__((ext_vector_type(8)))  float    v8f;
typedef __attribute__((ext_vector_type(4)))  float    f4;

#define NEG_SLOPE 0.01f

__device__ __forceinline__ float lrelu(float v) { return v >= 0.0f ? v : NEG_SLOPE * v; }

// ---------------------------------------------------------------------------
// Prepass: transpose W_conv (256x64 f32, row-major) -> Wt (64x256 f16)
// so WMMA B-fragment loads are contiguous along K.
// ---------------------------------------------------------------------------
__global__ void transpose_w_kernel(const float* __restrict__ W, _Float16* __restrict__ Wt) {
    int t = blockIdx.x * blockDim.x + threadIdx.x;   // 256*64 = 16384
    if (t >= 256 * 64) return;
    int k = t >> 6;          // 0..255
    int c = t & 63;          // 0..63
    Wt[c * 256 + k] = (_Float16)W[t];
}

// ---------------------------------------------------------------------------
// h = x @ W_conv via v_wmma_f32_16x16x32_f16.
// blockDim = 64 (2 waves). Each wave computes a 16x64 output tile:
// 4 accumulators (16x16 each), 8 K-steps of 32, A-fragment shared across tiles.
// grid.x = N/32 (N % 32 == 0 for N=100000).
// ---------------------------------------------------------------------------
__global__ void __launch_bounds__(64) gemm_xw_kernel(const float* __restrict__ x,
                                                     const _Float16* __restrict__ Wt,
                                                     float* __restrict__ h) {
    const int lane = threadIdx.x & 31;
    const int wave = threadIdx.x >> 5;
    const int hi   = lane >> 4;      // 0: K-low halves, 1: K+8 halves
    const int r    = lane & 15;      // row (A) / col (B,C,D)
    const int rowBase = (blockIdx.x * 2 + wave) * 16;

    v8f c0 = {}, c1 = {}, c2 = {}, c3 = {};

    const float*    xrow = x  + (size_t)(rowBase + r) * 256 + hi * 8;
    const _Float16* w0   = Wt + (size_t)(0 * 16 + r) * 256 + hi * 8;
    const _Float16* w1   = Wt + (size_t)(1 * 16 + r) * 256 + hi * 8;
    const _Float16* w2   = Wt + (size_t)(2 * 16 + r) * 256 + hi * 8;
    const _Float16* w3   = Wt + (size_t)(3 * 16 + r) * 256 + hi * 8;

    for (int kk = 0; kk < 8; ++kk) {
        const int kb = kk * 32;

        // A fragment: 16 f16 per lane; elements 0..7 = K kb+hi*8+0..7,
        // elements 8..15 = K kb+hi*8+16..23 (per CDNA5 16-bit A layout).
        f4 a0 = *(const f4*)(xrow + kb);
        f4 a1 = *(const f4*)(xrow + kb + 4);
        f4 a2 = *(const f4*)(xrow + kb + 16);
        f4 a3 = *(const f4*)(xrow + kb + 20);
        v16h a;
        #pragma unroll
        for (int i = 0; i < 4; ++i) {
            a[i]      = (_Float16)a0[i];
            a[4 + i]  = (_Float16)a1[i];
            a[8 + i]  = (_Float16)a2[i];
            a[12 + i] = (_Float16)a3[i];
        }

        // B fragments: lane holds column r, same K striping as A.
        h8 b0lo = *(const h8*)(w0 + kb); h8 b0hi = *(const h8*)(w0 + kb + 16);
        h8 b1lo = *(const h8*)(w1 + kb); h8 b1hi = *(const h8*)(w1 + kb + 16);
        h8 b2lo = *(const h8*)(w2 + kb); h8 b2hi = *(const h8*)(w2 + kb + 16);
        h8 b3lo = *(const h8*)(w3 + kb); h8 b3hi = *(const h8*)(w3 + kb + 16);
        v16h b0v, b1v, b2v, b3v;
        #pragma unroll
        for (int i = 0; i < 8; ++i) {
            b0v[i] = b0lo[i]; b0v[8 + i] = b0hi[i];
            b1v[i] = b1lo[i]; b1v[8 + i] = b1hi[i];
            b2v[i] = b2lo[i]; b2v[8 + i] = b2hi[i];
            b3v[i] = b3lo[i]; b3v[8 + i] = b3hi[i];
        }

        c0 = __builtin_amdgcn_wmma_f32_16x16x32_f16(false, a, false, b0v, (short)0, c0, false, false);
        c1 = __builtin_amdgcn_wmma_f32_16x16x32_f16(false, a, false, b1v, (short)0, c1, false, false);
        c2 = __builtin_amdgcn_wmma_f32_16x16x32_f16(false, a, false, b2v, (short)0, c2, false, false);
        c3 = __builtin_amdgcn_wmma_f32_16x16x32_f16(false, a, false, b3v, (short)0, c3, false, false);
    }

    // C/D layout: VGPR j, lanes 0-15 -> M=j, lanes 16-31 -> M=8+j, N=lane%16.
    #pragma unroll
    for (int j = 0; j < 8; ++j) {
        float* p = h + (size_t)(rowBase + j + hi * 8) * 64;
        p[r]      = c0[j];
        p[16 + r] = c1[j];
        p[32 + r] = c2[j];
        p[48 + r] = c3[j];
    }
}

// ---------------------------------------------------------------------------
// Degree: init to 1 (self-loop), then count edge destinations.
// ---------------------------------------------------------------------------
__global__ void deg_init_kernel(unsigned int* __restrict__ deg, int N) {
    int i = blockIdx.x * blockDim.x + threadIdx.x;
    if (i < N) deg[i] = 1u;
}

__global__ void deg_count_kernel(const int* __restrict__ dst, unsigned int* __restrict__ deg, int E) {
    int e = blockIdx.x * blockDim.x + threadIdx.x;
    if (e < E) atomicAdd(&deg[dst[e]], 1u);
}

__global__ void dinv_kernel(const unsigned int* __restrict__ deg, float* __restrict__ dinv, int N) {
    int i = blockIdx.x * blockDim.x + threadIdx.x;
    if (i < N) {
        float d = (float)deg[i];
        dinv[i] = d > 0.0f ? rsqrtf(d) : 0.0f;
    }
}

// ---------------------------------------------------------------------------
// embed init: self-loop contribution  embed[i] = h[i] * dinv[i]^2
// (also clears the 0xAA-poisoned workspace deterministically).
// ---------------------------------------------------------------------------
__global__ void embed_init_kernel(const float* __restrict__ h, const float* __restrict__ dinv,
                                  float* __restrict__ embed, int total) {
    int t = blockIdx.x * blockDim.x + threadIdx.x;
    if (t >= total) return;
    int i = t >> 6;
    float di = dinv[i];
    embed[t] = h[t] * di * di;
}

// ---------------------------------------------------------------------------
// Edge aggregation: embed[dst] += h[src] * dinv[src]*dinv[dst]
// One thread per (edge, channel): coalesced 256B gathers of h[src],
// fp32 hardware atomics into L2-resident embed (25.6 MB << 192 MB L2).
// ---------------------------------------------------------------------------
__global__ void scatter_edges_kernel(const float* __restrict__ h, const float* __restrict__ dinv,
                                     const int* __restrict__ src, const int* __restrict__ dst,
                                     float* __restrict__ embed, int E) {
    int t = blockIdx.x * blockDim.x + threadIdx.x;
    if (t >= E * 64) return;
    int e = t >> 6;
    int c = t & 63;
    int s = src[e];
    int d = dst[e];
    float norm = dinv[s] * dinv[d];
    unsafeAtomicAdd(&embed[(size_t)d * 64 + c], h[(size_t)s * 64 + c] * norm);
}

// ---------------------------------------------------------------------------
// MLP head: bias + leakyReLU applied at gather time; per-thread pair.
// ---------------------------------------------------------------------------
__global__ void mlp_kernel(const float* __restrict__ embed, const int* __restrict__ idx,
                           const float* __restrict__ b_conv,
                           const float* __restrict__ W1, const float* __restrict__ b1,
                           const float* __restrict__ W2, const float* __restrict__ b2,
                           float* __restrict__ out, int B) {
    int b = blockIdx.x * blockDim.x + threadIdx.x;
    if (b >= B) return;
    int i0 = idx[2 * b];
    int i1 = idx[2 * b + 1];
    const float* e0 = embed + (size_t)i0 * 64;
    const float* e1 = embed + (size_t)i1 * 64;

    float o1[16];
    #pragma unroll
    for (int j = 0; j < 16; ++j) o1[j] = b1[j];

    for (int c = 0; c < 64; ++c) {
        float v = lrelu(e0[c] + b_conv[c]);
        const float* wr = W1 + c * 16;
        #pragma unroll
        for (int j = 0; j < 16; ++j) o1[j] += v * wr[j];
    }
    for (int c = 0; c < 64; ++c) {
        float v = lrelu(e1[c] + b_conv[c]);
        const float* wr = W1 + (64 + c) * 16;
        #pragma unroll
        for (int j = 0; j < 16; ++j) o1[j] += v * wr[j];
    }

    float acc = b2[0];
    #pragma unroll
    for (int j = 0; j < 16; ++j) acc += lrelu(o1[j]) * W2[j];

    out[b] = 1.0f / (1.0f + __expf(-acc));
}

// ---------------------------------------------------------------------------
extern "C" void kernel_launch(void* const* d_in, const int* in_sizes, int n_in,
                              void* d_out, int out_size, void* d_ws, size_t ws_size,
                              hipStream_t stream) {
    const float* x        = (const float*)d_in[0];   // (N, 256)
    const int*   edge     = (const int*)d_in[1];     // (2, E)
    const int*   index    = (const int*)d_in[2];     // (B, 2)
    const float* W_conv   = (const float*)d_in[3];   // (256, 64)
    const float* b_conv   = (const float*)d_in[4];   // (64,)
    const float* W1       = (const float*)d_in[5];   // (128, 16)
    const float* b1       = (const float*)d_in[6];   // (16,)
    const float* W2       = (const float*)d_in[7];   // (16, 1)
    const float* b2       = (const float*)d_in[8];   // (1,)
    float*       out      = (float*)d_out;

    const int N = in_sizes[0] / 256;    // 100000
    const int E = in_sizes[1] / 2;      // 1600000
    const int B = in_sizes[2] / 2;      // 16384
    const int* src = edge;
    const int* dst = edge + E;

    // Workspace layout
    float*        h     = (float*)d_ws;                         // N*64 f32
    float*        embed = h + (size_t)N * 64;                   // N*64 f32
    unsigned int* deg   = (unsigned int*)(embed + (size_t)N * 64); // N u32
    float*        dinv  = (float*)(deg + N);                    // N f32
    _Float16*     Wt    = (_Float16*)(dinv + N);                // 64*256 f16

    // 1. Transpose W_conv to f16 (B-fragment friendly layout)
    transpose_w_kernel<<<(256 * 64 + 255) / 256, 256, 0, stream>>>(W_conv, Wt);

    // 2. h = x @ W_conv  (WMMA, 2 waves/block, 32 rows/block)
    gemm_xw_kernel<<<N / 32, 64, 0, stream>>>(x, Wt, h);

    // 3. Degrees (self-loop folded into init) -> dinv = rsqrt(deg)
    deg_init_kernel<<<(N + 255) / 256, 256, 0, stream>>>(deg, N);
    deg_count_kernel<<<(E + 255) / 256, 256, 0, stream>>>(dst, deg, E);
    dinv_kernel<<<(N + 255) / 256, 256, 0, stream>>>(deg, dinv, N);

    // 4. embed = self-loop term; then scatter-add all edges (fp32 HW atomics)
    embed_init_kernel<<<(N * 64 + 255) / 256, 256, 0, stream>>>(h, dinv, embed, N * 64);
    {
        long long total = (long long)E * 64;
        int blocks = (int)((total + 255) / 256);
        scatter_edges_kernel<<<blocks, 256, 0, stream>>>(h, dinv, src, dst, embed, E);
    }

    // 5. MLP head (bias + leakyReLU fused into the gather)
    mlp_kernel<<<(B + 255) / 256, 256, 0, stream>>>(embed, index, b_conv, W1, b1, W2, b2, out, B);
}
